// MAN_4105988735533
// MI455X (gfx1250) — matrix operand order repaired and checked by reading, not compiled, not run on hardware
//
#include <hip/hip_runtime.h>
#include <hip/hip_bf16.h>

typedef _Float16 v16h __attribute__((ext_vector_type(16)));
typedef _Float16 v8h  __attribute__((ext_vector_type(8)));
typedef float    v8f  __attribute__((ext_vector_type(8)));

#define D_IN  64
#define D_HID 128
#define D_OUT 64
#define NCLS  10
#define VOTE_PAD 20    // 8*20 = 160 ≡ 32 (mod 64): lane halves hit disjoint banks
#define WAVES 4
#define NSPLIT 8       // train-dimension split -> 4096 waves total
#define LOG2E 1.4426950408889634f

// ---------------------------------------------------------------------------
// Phase 1: embed + L2-normalize one row per block; emit f16 normalized rows.
// `scale` folds softmax's log2(e) into the test-side embeddings so the attend
// loop can use the raw base-2 v_exp_f32 with a single add (no pk_mul).
// ---------------------------------------------------------------------------
__global__ void embed_norm_kernel(const float* __restrict__ x,
                                  const float* __restrict__ W1,
                                  const float* __restrict__ b1,
                                  const float* __restrict__ W2,
                                  const float* __restrict__ b2,
                                  _Float16* __restrict__ out,
                                  float scale)
{
    __shared__ float xr[D_IN];
    __shared__ float hh[D_HID];
    __shared__ float red[D_OUT];
    const int t   = threadIdx.x;          // 128 threads
    const int row = blockIdx.x;

    if (t < D_IN) xr[t] = x[(size_t)row * D_IN + t];
    __syncthreads();

    float acc = b1[t];
#pragma unroll
    for (int k = 0; k < D_IN; ++k) acc = fmaf(xr[k], W1[k * D_HID + t], acc);
    hh[t] = fmaxf(acc, 0.0f);
    __syncthreads();

    float e = 0.0f;
    if (t < D_OUT) {
        e = b2[t];
#pragma unroll
        for (int k = 0; k < D_HID; ++k) e = fmaf(hh[k], W2[k * D_OUT + t], e);
        red[t] = e * e;
    }
    __syncthreads();
    for (int s = D_OUT / 2; s > 0; s >>= 1) {
        if (t < s) red[t] += red[t + s];
        __syncthreads();
    }
    const float inv = rsqrtf(fmaxf(red[0], 1e-16f)) * scale;
    if (t < D_OUT) out[(size_t)row * D_OUT + t] = (_Float16)(e * inv);
}

// ---------------------------------------------------------------------------
// Phase 2: fused sims -> softmax numerators -> one-hot votes.
// Each wave owns 16 test rows x one train slice (n_train/NSPLIT rows).
// sims tile = two v_wmma_f32_16x16x32_f16 k-steps (K=64).
// A is pre-scaled by log2e, so c = log2e * cos_sim and p = exp2(c - log2e)
// = exp(cos_sim - 1): argument bounded in [-2*log2e, 0], so the raw
// v_exp_f32 (no denormal guard) is exact enough and always in range.
// Partials (votes[16][10], l[16]) go to workspace; finalize sums the splits.
// ---------------------------------------------------------------------------
__global__ void attend_kernel(const _Float16* __restrict__ testn,
                              const _Float16* __restrict__ trainn,
                              const int*      __restrict__ y_train,
                              float*          __restrict__ votes_part, // [NSPLIT][B_test][NCLS]
                              float*          __restrict__ l_part,     // [NSPLIT][B_test]
                              int n_train, int B_test)
{
    __shared__ float votes[WAVES][16][VOTE_PAD];
    const int tid   = threadIdx.x;
    const int w     = tid >> 5;
    const int lane  = tid & 31;
    const int n     = lane & 15;   // col within tile (B/C) and row m (A)
    const int h     = lane >> 4;   // lane half
    const int r0    = (blockIdx.x * WAVES + w) * 16;
    const int split = blockIdx.y;
    const int slice = n_train / NSPLIT;
    const int jbeg  = split * slice;

    // zero this wave's vote accumulators (uniform -> EXEC stays full)
    for (int i = lane; i < 16 * VOTE_PAD; i += 32)
        votes[w][i / VOTE_PAD][i % VOTE_PAD] = 0.0f;
    __syncthreads();

    // A fragments (test rows r0..r0+15): contiguous K order matching B,
    // lane(m=n, h): K = h*16+0..15 in VGPR0-7.
    const _Float16* arow = testn + (size_t)(r0 + n) * D_OUT;
    v16h a0, a1;
    {
        const v8h lo0 = *(const v8h*)(arow + h * 16);
        const v8h hi0 = *(const v8h*)(arow + 8 + h * 16);
        const v8h lo1 = *(const v8h*)(arow + 32 + h * 16);
        const v8h hi1 = *(const v8h*)(arow + 40 + h * 16);
#pragma unroll
        for (int i = 0; i < 8; ++i) {
            a0[i] = lo0[i]; a0[i + 8] = hi0[i];
            a1[i] = lo1[i]; a1[i + 8] = hi1[i];
        }
    }

    float l[8];
#pragma unroll
    for (int v = 0; v < 8; ++v) l[v] = 0.0f;

    for (int j0 = jbeg; j0 < jbeg + slice; j0 += 16) {
        // B fragments: lane holds contiguous 16 K-values of column n
        // (B = train^T, so column n = train row j0+n, row-major contiguous).
        const _Float16* brow = trainn + (size_t)(j0 + n) * D_OUT + h * 16;
        const v16h b0  = *(const v16h*)(brow);
        const v16h b1v = *(const v16h*)(brow + 32);
        const int  y   = y_train[j0 + n];

        v8f c = {};
        c = __builtin_amdgcn_wmma_f32_16x16x32_f16(false, a0, false, b0,
                                                   (short)0, c, false, false);
        c = __builtin_amdgcn_wmma_f32_16x16x32_f16(false, a1, false, b1v,
                                                   (short)0, c, false, false);
#pragma unroll
        for (int v = 0; v < 8; ++v) {
            // raw v_exp_f32: argument in [-2*log2e, 0], no range guard needed
            const float p = __builtin_amdgcn_exp2f(c[v] - LOG2E);
            l[v] += p;
            atomicAdd(&votes[w][v + 8 * h][y], p);  // ds_add_f32
        }
    }

    // reduce row denominators across the 16-lane half (xor<16 stays in half)
#pragma unroll
    for (int v = 0; v < 8; ++v) {
        float s = l[v];
        s += __shfl_xor(s, 1, 32);
        s += __shfl_xor(s, 2, 32);
        s += __shfl_xor(s, 4, 32);
        s += __shfl_xor(s, 8, 32);
        l[v] = s;
    }
    __syncthreads();

    // write this slice's partials (each slot owned by exactly one wave)
    if (n == 0) {
#pragma unroll
        for (int v = 0; v < 8; ++v)
            l_part[(size_t)split * B_test + r0 + v + 8 * h] = l[v];
    }
    if (n < NCLS) {
#pragma unroll
        for (int v = 0; v < 8; ++v) {
            const int m = v + 8 * h;
            votes_part[((size_t)split * B_test + r0 + m) * NCLS + n] = votes[w][m][n];
        }
    }
}

// ---------------------------------------------------------------------------
// Phase 3: sum partials over splits and normalize.
// ---------------------------------------------------------------------------
__global__ void finalize_kernel(const float* __restrict__ votes_part,
                                const float* __restrict__ l_part,
                                float* __restrict__ out, int B_test)
{
    const int i = blockIdx.x * blockDim.x + threadIdx.x;
    if (i >= B_test) return;
    float lsum = 0.0f;
#pragma unroll
    for (int s = 0; s < NSPLIT; ++s) lsum += l_part[(size_t)s * B_test + i];
    const float inv = 1.0f / lsum;
#pragma unroll
    for (int c = 0; c < NCLS; ++c) {
        float v = 0.0f;
#pragma unroll
        for (int s = 0; s < NSPLIT; ++s)
            v += votes_part[((size_t)s * B_test + i) * NCLS + c];
        out[(size_t)i * NCLS + c] = v * inv;
    }
}

// ---------------------------------------------------------------------------
extern "C" void kernel_launch(void* const* d_in, const int* in_sizes, int n_in,
                              void* d_out, int out_size, void* d_ws, size_t ws_size,
                              hipStream_t stream)
{
    const float* X       = (const float*)d_in[0];
    const float* x_train = (const float*)d_in[1];
    const int*   y_train = (const int*)  d_in[2];
    const float* W1      = (const float*)d_in[3];
    const float* b1      = (const float*)d_in[4];
    const float* W2      = (const float*)d_in[5];
    const float* b2      = (const float*)d_in[6];
    float* out = (float*)d_out;

    const int B_test  = in_sizes[0] / D_IN;   // 8192
    const int B_train = in_sizes[1] / D_IN;   // 16384

    // workspace layout (all L2-resident, ~6.2 MB total)
    _Float16* testn      = (_Float16*)d_ws;                         // [B_test ][64] f16
    _Float16* trainn     = testn + (size_t)B_test * D_OUT;          // [B_train][64] f16
    float*    votes_part = (float*)(trainn + (size_t)B_train * D_OUT); // [NSPLIT][B_test][10]
    float*    l_part     = votes_part + (size_t)NSPLIT * B_test * NCLS; // [NSPLIT][B_test]

    embed_norm_kernel<<<B_test,  128, 0, stream>>>(X,       W1, b1, W2, b2, testn, LOG2E);
    embed_norm_kernel<<<B_train, 128, 0, stream>>>(x_train, W1, b1, W2, b2, trainn, 1.0f);

    dim3 grid(B_test / (16 * WAVES), NSPLIT);
    attend_kernel<<<grid, 32 * WAVES, 0, stream>>>(testn, trainn, y_train,
                                                   votes_part, l_part,
                                                   B_train, B_test);

    finalize_kernel<<<(B_test + 127) / 128, 128, 0, stream>>>(votes_part, l_part,
                                                              out, B_test);
}
